// InternImageBlock_84104049590499
// MI455X (gfx1250) — compile-verified, matrix operand order checked by reading, not run on hardware
//
#include <hip/hip_runtime.h>
#include <hip/hip_bf16.h>

// ---------------------------------------------------------------------------
// InternImage block for MI455X (gfx1250, wave32).
// Dense GEMMs -> v_wmma_f32_16x16x32_bf16 (f32 accumulate), 32x32 tile/wave.
// LN / softmax / depthwise conv / DCN gather -> VALU + LDS kernels.
// Downsample conv lowered to im2col + WMMA GEMM.
// ---------------------------------------------------------------------------

typedef __bf16 bf16_t;
typedef __attribute__((ext_vector_type(8)))  __bf16 v8bf;
typedef __attribute__((ext_vector_type(16))) __bf16 v16bf;
typedef __attribute__((ext_vector_type(8)))  float  v8f;

#define NTOK   4608   // 8*24*24
#define CH     128
#define GRP    8
#define CG     16
#define KPTS   27
#define HIDN   512
#define DD0    8
#define HH0    24
#define WW0    24
#define OHW    12     // (24+2-3)/2+1
#define NPOS   576    // 4*12*12
#define KDS    3456   // 27*128
#define CO     256

__device__ __forceinline__ float gelu_exact(float x) {
    return 0.5f * x * (1.0f + erff(x * 0.70710678118654752f));
}

// 16-bit WMMA fragment: lane r=l&15 is the row; hi=l>>4 selects K-halves
// {0..7,16..23} vs {8..15,24..31}. Two contiguous 16B loads per lane.
__device__ __forceinline__ v16bf load_frag(const bf16_t* __restrict__ p) {
    v8bf lo = *(const v8bf*)(p);
    v8bf hh = *(const v8bf*)(p + 16);
    v16bf f;
#pragma unroll
    for (int i = 0; i < 8; ++i) { f[i] = lo[i]; f[i + 8] = hh[i]; }
    return f;
}

// ---------------------------------------------------------------------------
// WMMA GEMM: out[M,N] = A[M,K](bf16) * Bt[N,K]^T(bf16) + bias.
// One wave computes a 32x32 output tile: 2 A-fragments x 2 B-fragments ->
// 4 WMMAs per K-step (K stepped by 32), halving fragment traffic vs 16x16.
// Optional exact-GELU, residual add, f32 and/or bf16 stores.
// M, N must be multiples of 32; K a multiple of 32 (true for all call sites).
// ---------------------------------------------------------------------------
__global__ __launch_bounds__(256)
void gemm_wmma_bf16(const bf16_t* __restrict__ A, const bf16_t* __restrict__ Bt,
                    const float* __restrict__ bias, const float* __restrict__ resid,
                    float* __restrict__ outF, bf16_t* __restrict__ outB,
                    int M, int N, int K, int doGelu)
{
    int wave  = blockIdx.x * (blockDim.x >> 5) + (threadIdx.x >> 5);
    int lane  = threadIdx.x & 31;
    int ntile = N >> 5;
    int total = (M >> 5) * ntile;
    if (wave >= total) return;                 // wave-uniform: EXEC stays all-1s

    int tm = (wave / ntile) << 5;
    int tn = (wave % ntile) << 5;
    int r  = lane & 15;
    int hi = lane >> 4;

    const bf16_t* arow0 = A  + (size_t)(tm + r)      * K + hi * 8;
    const bf16_t* arow1 = A  + (size_t)(tm + 16 + r) * K + hi * 8;
    const bf16_t* brow0 = Bt + (size_t)(tn + r)      * K + hi * 8;
    const bf16_t* brow1 = Bt + (size_t)(tn + 16 + r) * K + hi * 8;

    v8f acc00 = {}, acc01 = {}, acc10 = {}, acc11 = {};
    for (int k0 = 0; k0 < K; k0 += 32) {
        __builtin_prefetch(arow0 + k0 + 32, 0, 0);
        __builtin_prefetch(arow1 + k0 + 32, 0, 0);
        __builtin_prefetch(brow0 + k0 + 32, 0, 0);
        __builtin_prefetch(brow1 + k0 + 32, 0, 0);
        v16bf a0 = load_frag(arow0 + k0);
        v16bf a1 = load_frag(arow1 + k0);
        v16bf b0 = load_frag(brow0 + k0);
        v16bf b1 = load_frag(brow1 + k0);
        acc00 = __builtin_amdgcn_wmma_f32_16x16x32_bf16(false, a0, false, b0, (short)0, acc00, false, false);
        acc01 = __builtin_amdgcn_wmma_f32_16x16x32_bf16(false, a0, false, b1, (short)0, acc01, false, false);
        acc10 = __builtin_amdgcn_wmma_f32_16x16x32_bf16(false, a1, false, b0, (short)0, acc10, false, false);
        acc11 = __builtin_amdgcn_wmma_f32_16x16x32_bf16(false, a1, false, b1, (short)0, acc11, false, false);
    }

    // Epilogue. C/D layout: VGPR i, lanes<16 -> row tm+i, lanes>=16 -> row tm+i+8.
    int col0 = tn + r;
    int col1 = tn + 16 + r;
    float bv0 = bias ? bias[col0] : 0.0f;
    float bv1 = bias ? bias[col1] : 0.0f;
#pragma unroll
    for (int i = 0; i < 8; ++i) {
        int row0 = tm + i + hi * 8;
        int row1 = tm + 16 + i + hi * 8;
        float v00 = acc00[i] + bv0, v01 = acc01[i] + bv1;
        float v10 = acc10[i] + bv0, v11 = acc11[i] + bv1;
        if (doGelu) {
            v00 = gelu_exact(v00); v01 = gelu_exact(v01);
            v10 = gelu_exact(v10); v11 = gelu_exact(v11);
        }
        if (resid) {
            v00 += resid[(size_t)row0 * N + col0]; v01 += resid[(size_t)row0 * N + col1];
            v10 += resid[(size_t)row1 * N + col0]; v11 += resid[(size_t)row1 * N + col1];
        }
        if (outF) {
            outF[(size_t)row0 * N + col0] = v00; outF[(size_t)row0 * N + col1] = v01;
            outF[(size_t)row1 * N + col0] = v10; outF[(size_t)row1 * N + col1] = v11;
        }
        if (outB) {
            outB[(size_t)row0 * N + col0] = (bf16_t)v00; outB[(size_t)row0 * N + col1] = (bf16_t)v01;
            outB[(size_t)row1 * N + col0] = (bf16_t)v10; outB[(size_t)row1 * N + col1] = (bf16_t)v11;
        }
    }
}

// ---------------------------------------------------------------------------
// LayerNorm over channel dim (blockDim == C, power of two), optional GELU,
// dual f32/bf16 stores so the consumer GEMM reads bf16 directly.
// ---------------------------------------------------------------------------
__global__ void ln_kernel(const float* __restrict__ in, const float* __restrict__ w,
                          const float* __restrict__ b, float* __restrict__ outF,
                          bf16_t* __restrict__ outB, int C, int doGelu)
{
    __shared__ float red[256];
    int n = blockIdx.x, c = threadIdx.x;
    float x = in[(size_t)n * C + c];
    red[c] = x; __syncthreads();
    for (int s = C >> 1; s > 0; s >>= 1) { if (c < s) red[c] += red[c + s]; __syncthreads(); }
    float mu = red[0] * (1.0f / C);
    __syncthreads();
    float d = x - mu;
    red[c] = d * d; __syncthreads();
    for (int s = C >> 1; s > 0; s >>= 1) { if (c < s) red[c] += red[c + s]; __syncthreads(); }
    float var = red[0] * (1.0f / C);
    float y = d * rsqrtf(var + 1e-5f) * w[c] + b[c];
    if (doGelu) y = gelu_exact(y);
    if (outF) outF[(size_t)n * C + c] = y;
    if (outB) outB[(size_t)n * C + c] = (bf16_t)y;
}

// ---------------------------------------------------------------------------
// Depthwise 3x3x3 conv (pad 1, stride 1). One token per block, one channel
// per thread. w27: [C][27].
// ---------------------------------------------------------------------------
__global__ __launch_bounds__(CH)
void dwconv_kernel(const float* __restrict__ in, const float* __restrict__ w27,
                   const float* __restrict__ bias, float* __restrict__ out)
{
    int n = blockIdx.x, c = threadIdx.x;
    int ww = n % WW0, hh = (n / WW0) % HH0, dd = n / (WW0 * HH0);
    float acc = bias[c];
#pragma unroll
    for (int kd = 0; kd < 3; ++kd)
#pragma unroll
        for (int kh = 0; kh < 3; ++kh)
#pragma unroll
            for (int kw = 0; kw < 3; ++kw) {
                int id = dd + kd - 1, ih = hh + kh - 1, iw = ww + kw - 1;
                if ((unsigned)id < (unsigned)DD0 && (unsigned)ih < (unsigned)HH0 &&
                    (unsigned)iw < (unsigned)WW0)
                    acc += in[(((size_t)id * HH0 + ih) * WW0 + iw) * CH + c] *
                           w27[c * 27 + (kd * 3 + kh) * 3 + kw];
            }
    out[(size_t)n * CH + c] = acc;
}

// ---------------------------------------------------------------------------
// Offset (648) + mask-logit (216) projections for one token per block.
// z token staged in LDS; columns striped across 256 threads.
// ---------------------------------------------------------------------------
__global__ __launch_bounds__(256)
void offmask_kernel(const float* __restrict__ z,
                    const float* __restrict__ offw, const float* __restrict__ offb,
                    const float* __restrict__ mskw, const float* __restrict__ mskb,
                    float* __restrict__ off, float* __restrict__ msk)
{
    __shared__ float sz[CH];
    int n = blockIdx.x, t = threadIdx.x;
    if (t < CH) sz[t] = z[(size_t)n * CH + t];
    __syncthreads();
    for (int col = t; col < GRP * KPTS * 3; col += 256) {
        float a = offb[col];
        for (int c = 0; c < CH; ++c) a += sz[c] * offw[c * (GRP * KPTS * 3) + col];
        off[(size_t)n * (GRP * KPTS * 3) + col] = a;
    }
    for (int col = t; col < GRP * KPTS; col += 256) {
        float a = mskb[col];
        for (int c = 0; c < CH; ++c) a += sz[c] * mskw[c * (GRP * KPTS) + col];
        msk[(size_t)n * (GRP * KPTS) + col] = a;
    }
}

// Softmax over K=27 per (token, group), in place.
__global__ void softmax_k_kernel(float* __restrict__ msk)
{
    int i = blockIdx.x * blockDim.x + threadIdx.x;
    if (i >= NTOK * GRP) return;
    float* p = msk + (size_t)(i >> 3) * (GRP * KPTS) + (i & 7) * KPTS;
    float mx = p[0];
    for (int k = 1; k < KPTS; ++k) mx = fmaxf(mx, p[k]);
    float s = 0.0f, e[KPTS];
    for (int k = 0; k < KPTS; ++k) { e[k] = expf(p[k] - mx); s += e[k]; }
    float inv = 1.0f / s;
    for (int k = 0; k < KPTS; ++k) p[k] = e[k] * inv;
}

// ---------------------------------------------------------------------------
// DCNv3 core. One token per block (128 threads = G*Cg). Phase 1: 216 (g,k)
// tasks cooperatively compute the 8 trilinear corner {index, weight*mask}
// into LDS. Phase 2: thread (g,c) accumulates 27*8 gathered values.
// pix = ref*size + gridpoint + off - 0.5 ; ref*size = 1.5 + i*(s-3)/(s-1).
// ---------------------------------------------------------------------------
__global__ __launch_bounds__(CH)
void dcn_kernel(const float* __restrict__ v, const float* __restrict__ off,
                const float* __restrict__ msk, bf16_t* __restrict__ out)
{
    __shared__ int   s_idx[GRP * KPTS * 8];
    __shared__ float s_w  [GRP * KPTS * 8];
    int n = blockIdx.x, t = threadIdx.x;
    int ww = n % WW0, hh = (n / WW0) % HH0, dd = n / (WW0 * HH0);

    const float sd = (DD0 - 3.0f) / (DD0 - 1.0f);   // 5/7
    const float sh = (HH0 - 3.0f) / (HH0 - 1.0f);   // 21/23
    const float sw = (WW0 - 3.0f) / (WW0 - 1.0f);

    for (int q = t; q < GRP * KPTS; q += CH) {
        int k  = q % KPTS;
        int kd = k / 9, kh = (k / 3) % 3, kw = k % 3;
        const float* o = off + ((size_t)n * GRP * KPTS + q) * 3;
        float pd = 1.0f + dd * sd + (float)(kd - 1) + o[0];
        float ph = 1.0f + hh * sh + (float)(kh - 1) + o[1];
        float pw = 1.0f + ww * sw + (float)(kw - 1) + o[2];
        float fd = floorf(pd), fh = floorf(ph), fw = floorf(pw);
        float rd = pd - fd, rh = ph - fh, rw = pw - fw;
        int id = (int)fd, ih = (int)fh, iw = (int)fw;
        float m = msk[(size_t)n * GRP * KPTS + q];
#pragma unroll
        for (int cc = 0; cc < 8; ++cc) {
            int cd = (cc >> 2) & 1, chh = (cc >> 1) & 1, cww = cc & 1;
            int jd = id + cd, jh = ih + chh, jw = iw + cww;
            float wt = (cd ? rd : 1.0f - rd) * (chh ? rh : 1.0f - rh) *
                       (cww ? rw : 1.0f - rw);
            bool valid = (unsigned)jd < (unsigned)DD0 && (unsigned)jh < (unsigned)HH0 &&
                         (unsigned)jw < (unsigned)WW0;
            int cjd = jd < 0 ? 0 : (jd > DD0 - 1 ? DD0 - 1 : jd);
            int cjh = jh < 0 ? 0 : (jh > HH0 - 1 ? HH0 - 1 : jh);
            int cjw = jw < 0 ? 0 : (jw > WW0 - 1 ? WW0 - 1 : jw);
            s_idx[q * 8 + cc] = (cjd * HH0 + cjh) * WW0 + cjw;
            s_w  [q * 8 + cc] = valid ? wt * m : 0.0f;
        }
    }
    __syncthreads();

    int g = t >> 4, c = t & 15;
    int chan = g * CG + c;
    float acc = 0.0f;
    for (int k = 0; k < KPTS; ++k) {
        int q = g * KPTS + k;
#pragma unroll
        for (int cc = 0; cc < 8; ++cc)
            acc += s_w[q * 8 + cc] * v[(size_t)s_idx[q * 8 + cc] * CH + chan];
    }
    out[(size_t)n * CH + chan] = (bf16_t)acc;
}

// ---------------------------------------------------------------------------
// im2col for the stride-2 pad-1 3x3x3 downsample conv: [576][27*128] bf16.
// ---------------------------------------------------------------------------
__global__ __launch_bounds__(CH)
void im2col_kernel(const float* __restrict__ in, bf16_t* __restrict__ col)
{
    int pos = blockIdx.x, ci = threadIdx.x;
    int ow = pos % OHW, oh = (pos / OHW) % OHW, od = pos / (OHW * OHW);
    for (int tap = 0; tap < 27; ++tap) {
        int kd = tap / 9, kh = (tap / 3) % 3, kw = tap % 3;
        int id = od * 2 - 1 + kd, ih = oh * 2 - 1 + kh, iw = ow * 2 - 1 + kw;
        float val = 0.0f;
        if ((unsigned)id < (unsigned)DD0 && (unsigned)ih < (unsigned)HH0 &&
            (unsigned)iw < (unsigned)WW0)
            val = in[(((size_t)id * HH0 + ih) * WW0 + iw) * CH + ci];
        col[(size_t)pos * KDS + tap * CH + ci] = (bf16_t)val;
    }
}

// Convert f32 weight [K][N] -> bf16 transposed [N][K].
__global__ void cvt_wt_kernel(const float* __restrict__ w, bf16_t* __restrict__ wt,
                              int K, int N)
{
    int i = blockIdx.x * blockDim.x + threadIdx.x;
    if (i >= K * N) return;
    int k = i / N, nn = i % N;
    wt[(size_t)nn * K + k] = (bf16_t)w[i];
}

// ds_w [256][128][27] -> bf16 [256][3456] with k = tap*128 + ci (im2col order).
__global__ void cvt_dsw_kernel(const float* __restrict__ w, bf16_t* __restrict__ wt)
{
    int i = blockIdx.x * blockDim.x + threadIdx.x;
    if (i >= CO * CH * 27) return;
    int co = i / (CH * 27), rem = i % (CH * 27);
    int ci = rem / 27, tap = rem % 27;
    wt[(size_t)co * KDS + tap * CH + ci] = (bf16_t)w[i];
}

// ---------------------------------------------------------------------------
// Host orchestration.
// ---------------------------------------------------------------------------
static inline void gemm_launch(hipStream_t s, const bf16_t* A, const bf16_t* Bt,
                               const float* bias, const float* resid,
                               float* outF, bf16_t* outB, int M, int N, int K, int gelu)
{
    int tiles  = (M / 32) * (N / 32);          // one wave per 32x32 tile
    int blocks = (tiles + 7) / 8;              // 8 waves / 256-thread block
    gemm_wmma_bf16<<<blocks, 256, 0, s>>>(A, Bt, bias, resid, outF, outB, M, N, K, gelu);
}

extern "C" void kernel_launch(void* const* d_in, const int* in_sizes, int n_in,
                              void* d_out, int out_size, void* d_ws, size_t ws_size,
                              hipStream_t stream)
{
    (void)in_sizes; (void)n_in; (void)out_size; (void)ws_size;
    const float* x_in    = (const float*)d_in[0];
    const float* norm1_w = (const float*)d_in[1];
    const float* norm1_b = (const float*)d_in[2];
    const float* inp_w   = (const float*)d_in[3];
    const float* inp_b   = (const float*)d_in[4];
    const float* dw_w    = (const float*)d_in[5];
    const float* dw_b    = (const float*)d_in[6];
    const float* dwn_w   = (const float*)d_in[7];
    const float* dwn_b   = (const float*)d_in[8];
    const float* off_w   = (const float*)d_in[9];
    const float* off_b   = (const float*)d_in[10];
    const float* msk_w   = (const float*)d_in[11];
    const float* msk_b   = (const float*)d_in[12];
    const float* outp_w  = (const float*)d_in[13];
    const float* outp_b  = (const float*)d_in[14];
    const float* norm2_w = (const float*)d_in[15];
    const float* norm2_b = (const float*)d_in[16];
    const float* fc1_w   = (const float*)d_in[17];
    const float* fc1_b   = (const float*)d_in[18];
    const float* fc2_w   = (const float*)d_in[19];
    const float* fc2_b   = (const float*)d_in[20];
    const float* fn_w    = (const float*)d_in[21];
    const float* fn_b    = (const float*)d_in[22];
    const float* ds_w    = (const float*)d_in[23];
    const float* ds_ln_w = (const float*)d_in[24];
    const float* ds_ln_b = (const float*)d_in[25];

    char* ws = (char*)d_ws;
    size_t o = 0;
    auto carve = [&](size_t bytes) { char* p = ws + o; o += (bytes + 255) & ~(size_t)255; return p; };

    float*  x      = (float*) carve((size_t)NTOK * CH * 4);
    float*  lnxF   = (float*) carve((size_t)NTOK * CH * 4);
    bf16_t* lnxB   = (bf16_t*)carve((size_t)NTOK * CH * 2);
    float*  vF     = (float*) carve((size_t)NTOK * CH * 4);
    float*  z      = (float*) carve((size_t)NTOK * CH * 4);
    float*  z2     = (float*) carve((size_t)NTOK * CH * 4);
    float*  offb   = (float*) carve((size_t)NTOK * GRP * KPTS * 3 * 4);
    float*  mskb   = (float*) carve((size_t)NTOK * GRP * KPTS * 4);
    bf16_t* dcnB   = (bf16_t*)carve((size_t)NTOK * CH * 2);
    bf16_t* hB     = (bf16_t*)carve((size_t)NTOK * HIDN * 2);
    bf16_t* inpwt  = (bf16_t*)carve((size_t)2 * CH * CH * 2);
    bf16_t* outpwt = (bf16_t*)carve((size_t)2 * CH * CH * 2);
    bf16_t* fc1wt  = (bf16_t*)carve((size_t)2 * CH * HIDN * 2);
    bf16_t* fc2wt  = (bf16_t*)carve((size_t)2 * HIDN * CH * 2);
    bf16_t* dswt   = (bf16_t*)carve((size_t)CO * KDS * 2);
    bf16_t* colB   = (bf16_t*)carve((size_t)NPOS * KDS * 2);
    float*  yds    = (float*) carve((size_t)NPOS * CO * 4);

    // Working copy of x (inputs must stay untouched).
    hipMemcpyAsync(x, x_in, (size_t)NTOK * CH * 4, hipMemcpyDeviceToDevice, stream);

    // Pre-convert / transpose all GEMM weights to bf16 [N][K].
    for (int l = 0; l < 2; ++l) {
        cvt_wt_kernel<<<(CH * CH + 255) / 256, 256, 0, stream>>>(
            inp_w + (size_t)l * CH * CH, inpwt + (size_t)l * CH * CH, CH, CH);
        cvt_wt_kernel<<<(CH * CH + 255) / 256, 256, 0, stream>>>(
            outp_w + (size_t)l * CH * CH, outpwt + (size_t)l * CH * CH, CH, CH);
        cvt_wt_kernel<<<(CH * HIDN + 255) / 256, 256, 0, stream>>>(
            fc1_w + (size_t)l * CH * HIDN, fc1wt + (size_t)l * CH * HIDN, CH, HIDN);
        cvt_wt_kernel<<<(HIDN * CH + 255) / 256, 256, 0, stream>>>(
            fc2_w + (size_t)l * HIDN * CH, fc2wt + (size_t)l * HIDN * CH, HIDN, CH);
    }
    cvt_dsw_kernel<<<(CO * CH * 27 + 255) / 256, 256, 0, stream>>>(ds_w, dswt);

    for (int l = 0; l < 2; ++l) {
        // 1) ln(x) -> f32 (for dwconv) + bf16 (for input-proj GEMM)
        ln_kernel<<<NTOK, CH, 0, stream>>>(x, norm1_w + l * CH, norm1_b + l * CH,
                                           lnxF, lnxB, CH, 0);
        // 2) v = ln(x) @ inp_w + inp_b   (WMMA)
        gemm_launch(stream, lnxB, inpwt + (size_t)l * CH * CH, inp_b + l * CH,
                    nullptr, vF, nullptr, NTOK, CH, CH, 0);
        // 3) depthwise 3x3x3 conv on ln(x)
        dwconv_kernel<<<NTOK, CH, 0, stream>>>(lnxF, dw_w + (size_t)l * CH * 27,
                                               dw_b + l * CH, z);
        // 4) z2 = gelu(ln(z))
        ln_kernel<<<NTOK, CH, 0, stream>>>(z, dwn_w + l * CH, dwn_b + l * CH,
                                           z2, nullptr, CH, 1);
        // 5) offset / mask projections + 6) softmax over K
        offmask_kernel<<<NTOK, 256, 0, stream>>>(
            z2, off_w + (size_t)l * CH * GRP * KPTS * 3, off_b + (size_t)l * GRP * KPTS * 3,
            msk_w + (size_t)l * CH * GRP * KPTS, msk_b + (size_t)l * GRP * KPTS, offb, mskb);
        softmax_k_kernel<<<(NTOK * GRP + 255) / 256, 256, 0, stream>>>(mskb);
        // 7) deformable trilinear gather -> bf16
        dcn_kernel<<<NTOK, CH, 0, stream>>>(vF, offb, mskb, dcnB);
        // 8) x = x + dcn @ outp_w + outp_b   (WMMA, fused residual)
        gemm_launch(stream, dcnB, outpwt + (size_t)l * CH * CH, outp_b + l * CH,
                    x, x, nullptr, NTOK, CH, CH, 0);
        // 9) MLP: h = gelu(ln2(x) @ fc1 + b) ; x = x + h @ fc2 + b
        ln_kernel<<<NTOK, CH, 0, stream>>>(x, norm2_w + l * CH, norm2_b + l * CH,
                                           nullptr, lnxB, CH, 0);
        gemm_launch(stream, lnxB, fc1wt + (size_t)l * CH * HIDN, fc1_b + l * HIDN,
                    nullptr, nullptr, hB, NTOK, HIDN, CH, 1);
        gemm_launch(stream, hB, fc2wt + (size_t)l * HIDN * CH, fc2_b + l * CH,
                    x, x, nullptr, NTOK, CH, HIDN, 0);
    }

    // Final norm -> im2col -> downsample GEMM (no bias) -> LN -> d_out
    ln_kernel<<<NTOK, CH, 0, stream>>>(x, fn_w, fn_b, lnxF, nullptr, CH, 0);
    im2col_kernel<<<NPOS, CH, 0, stream>>>(lnxF, colB);
    gemm_launch(stream, colB, dswt, nullptr, nullptr, yds, nullptr, NPOS, CO, KDS, 0);
    ln_kernel<<<NPOS, CO, 0, stream>>>(yds, ds_ln_w, ds_ln_b, (float*)d_out, nullptr, CO, 0);
}
